// Head_74028056313973
// MI455X (gfx1250) — matrix operand order repaired and checked by reading, not compiled, not run on hardware
//
#include <hip/hip_runtime.h>
#include <hip/hip_bf16.h>

typedef __attribute__((ext_vector_type(16))) _Float16 v16h;
typedef __attribute__((ext_vector_type(8)))  _Float16 v8h;
typedef __attribute__((ext_vector_type(8)))  float    v8f;
typedef __attribute__((ext_vector_type(4)))  int      i32x4;

#define B_  4
#define S_  2048
#define D_  1024
#define H_  16
#define HD_ 64

#define M_TOT (B_ * S_)   // 8192
#define N_TOT (3 * D_)    // 3072

// CDNA5 async global->LDS path (guarded: falls back to plain LDS stores)
#if defined(__has_builtin)
#if __has_builtin(__builtin_amdgcn_global_load_async_to_lds_b128)
#define HAVE_ASYNC_LDS 1
#endif
#endif

#ifdef HAVE_ASYNC_LDS
typedef __attribute__((address_space(1))) i32x4* g_b128_p;   // global int4*
typedef __attribute__((address_space(3))) i32x4* l_b128_p;   // LDS int4*
#endif

// ---------------------------------------------------------------------------
// Kernel 0: fp32 -> f16 conversion (once, so GEMM hot loop has no v_cvt).
// ---------------------------------------------------------------------------
__global__ __launch_bounds__(256) void cvt_f16_kernel(
    const float* __restrict__ in, _Float16* __restrict__ out, int n8)
{
  const int i = blockIdx.x * 256 + threadIdx.x;
  if (i < n8) {
    const v8f f = *(const v8f*)(in + (size_t)i * 8);
    v8h h;
    #pragma unroll
    for (int j = 0; j < 8; ++j) h[j] = (_Float16)f[j];
    *(v8h*)(out + (size_t)i * 8) = h;
  }
}

// ---------------------------------------------------------------------------
// Kernel 1: fused QKV projection as LDS-blocked GEMM with a double-buffered
// async global->LDS pipeline.  Block tile 128x128, BK=32, 8 waves, each wave
// owns 32x64 (2x4 WMMA tiles) -> 8 v_wmma per K-step.
// Epilogue is fully uniform: a 128-col N-block lies in one projection, a
// 128-row M-block lies in one batch.  V is stored transposed [B,H,HD,S].
// ---------------------------------------------------------------------------
#define BM 128
#define BN 128
#define BK 32
#define LROW 40   // padded LDS row (halves): 80B stride, conflict-free + 16B aligned

__global__ __launch_bounds__(256) void qkv_gemm_kernel(
    const _Float16* __restrict__ xh,
    const _Float16* __restrict__ wh,
    _Float16* __restrict__ Qh,
    _Float16* __restrict__ Kh,
    _Float16* __restrict__ Vt)
{
  __shared__ __attribute__((aligned(32))) _Float16 xs[2][BM][LROW];
  __shared__ __attribute__((aligned(32))) _Float16 wsb[2][BN][LROW];

  const int tid  = threadIdx.x;
  const int lane = tid & 31;
  const int wave = tid >> 5;
  const int hi = lane >> 4, ln = lane & 15;
  const int wm = wave >> 1;          // 0..3 : 32-row strip
  const int wn = wave & 1;           // 0..1 : 64-col strip

  const int NB = N_TOT / BN;         // 24
  const int m0 = (blockIdx.x / NB) * BM;
  const int n0 = (blockIdx.x % NB) * BN;

  // staging: each of 256 threads moves 16 halves of x and 16 halves of w
  const int sr   = tid >> 1;         // 0..127
  const int scol = (tid & 1) * 16;   // 0 or 16

#ifdef HAVE_ASYNC_LDS
#define STAGE(K0S, BUF) do {                                                       \
    const _Float16* gx_ = xh + (size_t)(m0 + sr) * D_ + (K0S) + scol;              \
    const _Float16* gw_ = wh + (size_t)(n0 + sr) * D_ + (K0S) + scol;              \
    __builtin_amdgcn_global_load_async_to_lds_b128((g_b128_p)gx_,       (l_b128_p)&xs[BUF][sr][scol],      0, 0); \
    __builtin_amdgcn_global_load_async_to_lds_b128((g_b128_p)(gx_ + 8), (l_b128_p)&xs[BUF][sr][scol + 8],  0, 0); \
    __builtin_amdgcn_global_load_async_to_lds_b128((g_b128_p)gw_,       (l_b128_p)&wsb[BUF][sr][scol],     0, 0); \
    __builtin_amdgcn_global_load_async_to_lds_b128((g_b128_p)(gw_ + 8), (l_b128_p)&wsb[BUF][sr][scol + 8], 0, 0); \
  } while (0)
#else
#define STAGE(K0S, BUF) do {                                                       \
    const _Float16* gx_ = xh + (size_t)(m0 + sr) * D_ + (K0S) + scol;              \
    const _Float16* gw_ = wh + (size_t)(n0 + sr) * D_ + (K0S) + scol;              \
    *(v8h*)&xs[BUF][sr][scol]      = *(const v8h*)gx_;                             \
    *(v8h*)&xs[BUF][sr][scol + 8]  = *(const v8h*)(gx_ + 8);                       \
    *(v8h*)&wsb[BUF][sr][scol]     = *(const v8h*)gw_;                             \
    *(v8h*)&wsb[BUF][sr][scol + 8] = *(const v8h*)(gw_ + 8);                       \
  } while (0)
#endif

  v8f acc[2][4] = {};

  STAGE(0, 0);                       // prologue: fill buffer 0

  for (int k0 = 0; k0 < D_; k0 += BK) {
    const int cur = (k0 / BK) & 1;
    const int nxt = cur ^ 1;
    if (k0 + BK < D_) {
      STAGE(k0 + BK, nxt);           // prefetch next slice into other buffer
#ifdef HAVE_ASYNC_LDS
      __builtin_amdgcn_s_wait_asynccnt(4);   // in-order: current slice landed
#endif
    } else {
#ifdef HAVE_ASYNC_LDS
      __builtin_amdgcn_s_wait_asynccnt(0);
#endif
    }
    __syncthreads();                 // current buffer visible to all waves

    // A fragments: rows wm*32 + sm*16 + ln, K striped per half-wave
    v16h a[2];
    #pragma unroll
    for (int sm = 0; sm < 2; ++sm) {
      const _Float16* row = &xs[cur][wm * 32 + sm * 16 + ln][0];
      const v8h p0 = *(const v8h*)(row + hi * 16);
      const v8h p1 = *(const v8h*)(row + hi * 16 + 8);
      #pragma unroll
      for (int i = 0; i < 8; ++i) { a[sm][i] = p0[i]; a[sm][8 + i] = p1[i]; }
    }
    // B fragments: cols wn*64 + sn*16 + ln; lanes<16 K 0..15, lanes>=16 K 16..31
    v16h bf[4];
    #pragma unroll
    for (int sn = 0; sn < 4; ++sn) {
      const _Float16* row = &wsb[cur][wn * 64 + sn * 16 + ln][0];
      const v8h p0 = *(const v8h*)(row + hi * 16);
      const v8h p1 = *(const v8h*)(row + hi * 16 + 8);
      #pragma unroll
      for (int i = 0; i < 8; ++i) { bf[sn][i] = p0[i]; bf[sn][8 + i] = p1[i]; }
    }
    #pragma unroll
    for (int sm = 0; sm < 2; ++sm)
      #pragma unroll
      for (int sn = 0; sn < 4; ++sn)
        acc[sm][sn] = __builtin_amdgcn_wmma_f32_16x16x32_f16(
            false, a[sm], false, bf[sn], (short)0, acc[sm][sn], false, false);

    __syncthreads();                 // all reads done before buffer reuse
  }
#undef STAGE

  // ---- uniform epilogue ---------------------------------------------------
  const int proj = n0 >> 10;                      // uniform per block
  const int hb   = ((n0 & (D_ - 1)) >> 6) + wn;   // head index, uniform per wave
  const int bb   = m0 >> 11;                      // batch, uniform per block
  const int sb   = (m0 & (S_ - 1)) + wm * 32;     // seq base, uniform per wave

  if (proj == 2) {
    // V transposed [B,H,HD,S]: consecutive v <-> consecutive s -> 16B stores
    #pragma unroll
    for (int sn = 0; sn < 4; ++sn) {
      _Float16* vrow = Vt + (((size_t)bb * H_ + hb) * HD_ + sn * 16 + ln) * S_;
      #pragma unroll
      for (int sm = 0; sm < 2; ++sm) {
        v8h pk;
        #pragma unroll
        for (int v = 0; v < 8; ++v) pk[v] = (_Float16)acc[sm][sn][v];
        *(v8h*)(vrow + sb + sm * 16 + hi * 8) = pk;
      }
    }
  } else {
    _Float16* base = (proj ? Kh : Qh) + ((size_t)bb * H_ + hb) * S_ * HD_;
    #pragma unroll
    for (int sn = 0; sn < 4; ++sn) {
      const int e = sn * 16 + ln;
      #pragma unroll
      for (int sm = 0; sm < 2; ++sm)
        #pragma unroll
        for (int v = 0; v < 8; ++v)
          base[(size_t)(sb + sm * 16 + v + hi * 8) * HD_ + e] = (_Float16)acc[sm][sn][v];
    }
  }
}

// ---------------------------------------------------------------------------
// Kernel 2: causal flash attention, one wave per 32-query tile.
// KV loop split: maskless full blocks + one masked diagonal block.
// ---------------------------------------------------------------------------
template <bool MASK>
__device__ __forceinline__ void fa_kv_block(
    const _Float16* __restrict__ Kb, const _Float16* __restrict__ Vtb,
    _Float16 (* __restrict__ pslab)[40],        // per-wave [32][40] LDS slab
    const v16h aq[2][2], v8f acc[2][4], float mrow[2][8], float lrow[2][8],
    int kv0, int q0, int hi, int ln)
{
  const float scale = 0.125f;   // 1/sqrt(HD)

  // K fragments for this 32-key block (shared by both M-subtiles)
  v16h bk[2][2];
  #pragma unroll
  for (int f = 0; f < 2; ++f) {
    const _Float16* krow = Kb + (size_t)(kv0 + f * 16 + ln) * HD_;
    #pragma unroll
    for (int ck = 0; ck < 2; ++ck)
      bk[f][ck] = *(const v16h*)(krow + ck * 32 + hi * 16);
  }
  // scores: 2 sm x 2 f tiles, accumulating over 2 K-chunks
  v8f sc[2][2];
  #pragma unroll
  for (int sm = 0; sm < 2; ++sm)
    #pragma unroll
    for (int f = 0; f < 2; ++f) {
      v8f cf = {};
      #pragma unroll
      for (int ck = 0; ck < 2; ++ck)
        cf = __builtin_amdgcn_wmma_f32_16x16x32_f16(
            false, aq[sm][ck], false, bk[f][ck], (short)0, cf, false, false);
      sc[sm][f] = cf;
    }

  // scale (+ causal mask only on the diagonal block) + online softmax
  #pragma unroll
  for (int sm = 0; sm < 2; ++sm) {
    #pragma unroll
    for (int f = 0; f < 2; ++f) {
      #pragma unroll
      for (int v = 0; v < 8; ++v) {
        float sv = sc[sm][f][v] * scale;
        if (MASK) {
          const int key = kv0 + f * 16 + ln;
          const int qr  = q0 + sm * 16 + v + hi * 8;
          sv = (key <= qr) ? sv : -1e30f;
        }
        sc[sm][f][v] = sv;
      }
    }
    #pragma unroll
    for (int v = 0; v < 8; ++v) {
      float rmax = fmaxf(sc[sm][0][v], sc[sm][1][v]);
      #pragma unroll
      for (int off = 1; off < 16; off <<= 1)
        rmax = fmaxf(rmax, __shfl_xor(rmax, off, 32));
      const float mnew = fmaxf(mrow[sm][v], rmax);
      const float corr = __expf(mrow[sm][v] - mnew);
      const float p0 = __expf(sc[sm][0][v] - mnew);
      const float p1 = __expf(sc[sm][1][v] - mnew);
      sc[sm][0][v] = p0; sc[sm][1][v] = p1;
      float rsum = p0 + p1;
      #pragma unroll
      for (int off = 1; off < 16; off <<= 1)
        rsum += __shfl_xor(rsum, off, 32);
      lrow[sm][v] = lrow[sm][v] * corr + rsum;
      mrow[sm][v] = mnew;
      #pragma unroll
      for (int t = 0; t < 4; ++t) acc[sm][t][v] *= corr;
    }
  }

  // P: C layout -> per-wave LDS slab (row-major 32x32) -> A fragments
  #pragma unroll
  for (int sm = 0; sm < 2; ++sm)
    #pragma unroll
    for (int f = 0; f < 2; ++f)
      #pragma unroll
      for (int v = 0; v < 8; ++v)
        pslab[sm * 16 + v + hi * 8][f * 16 + ln] = (_Float16)sc[sm][f][v];
  asm volatile("s_wait_dscnt 0" ::: "memory");

  v16h ap[2];
  #pragma unroll
  for (int sm = 0; sm < 2; ++sm) {
    const _Float16* prow = &pslab[sm * 16 + ln][0];
    const v8h p0 = *(const v8h*)(prow + hi * 16);
    const v8h p1 = *(const v8h*)(prow + hi * 16 + 8);
    #pragma unroll
    for (int i = 0; i < 8; ++i) { ap[sm][i] = p0[i]; ap[sm][8 + i] = p1[i]; }
  }
  asm volatile("" ::: "memory");

  // O += P * V : V fragments contiguous from transposed layout
  #pragma unroll
  for (int t = 0; t < 4; ++t) {
    const v16h bv = *(const v16h*)(Vtb + (size_t)(t * 16 + ln) * S_ + kv0 + hi * 16);
    #pragma unroll
    for (int sm = 0; sm < 2; ++sm)
      acc[sm][t] = __builtin_amdgcn_wmma_f32_16x16x32_f16(
          false, ap[sm], false, bv, (short)0, acc[sm][t], false, false);
  }
}

__global__ __launch_bounds__(128) void flash_attn_kernel(
    const _Float16* __restrict__ Qh,
    const _Float16* __restrict__ Kh,
    const _Float16* __restrict__ Vt,
    float* __restrict__ out)
{
  __shared__ __attribute__((aligned(32))) _Float16 pbuf[4][32][40];

  const int lane = threadIdx.x & 31;
  const int wave = threadIdx.x >> 5;
  const int hi = lane >> 4, ln = lane & 15;

  const int QT = S_ / 32;                 // 64 query tiles per head
  const int id = blockIdx.x * 4 + wave;
  const int qt = id % QT;
  const int h  = (id / QT) % H_;
  const int b  = id / (QT * H_);
  const int q0 = qt * 32;

  const _Float16* Qb  = Qh + (((size_t)b * H_ + h) * S_) * HD_;
  const _Float16* Kb  = Kh + (((size_t)b * H_ + h) * S_) * HD_;
  const _Float16* Vtb = Vt + (((size_t)b * H_ + h) * HD_) * S_;

  // Q fragments: [sm][ck], rows q0 + sm*16 + ln, HD=64 split into 2 K-chunks
  v16h aq[2][2];
  #pragma unroll
  for (int sm = 0; sm < 2; ++sm) {
    const _Float16* qrow = Qb + (size_t)(q0 + sm * 16 + ln) * HD_;
    #pragma unroll
    for (int ck = 0; ck < 2; ++ck) {
      const int kb = ck * 32 + hi * 16;
      const v8h p0 = *(const v8h*)(qrow + kb);
      const v8h p1 = *(const v8h*)(qrow + kb + 8);
      #pragma unroll
      for (int i = 0; i < 8; ++i) { aq[sm][ck][i] = p0[i]; aq[sm][ck][8 + i] = p1[i]; }
    }
  }

  float mrow[2][8], lrow[2][8];
  v8f acc[2][4] = {};
  #pragma unroll
  for (int sm = 0; sm < 2; ++sm)
    #pragma unroll
    for (int v = 0; v < 8; ++v) { mrow[sm][v] = -1e30f; lrow[sm][v] = 0.f; }

  // full (maskless) blocks, then the masked diagonal block
  for (int kv0 = 0; kv0 < q0; kv0 += 32)
    fa_kv_block<false>(Kb, Vtb, pbuf[wave], aq, acc, mrow, lrow, kv0, q0, hi, ln);
  fa_kv_block<true>(Kb, Vtb, pbuf[wave], aq, acc, mrow, lrow, q0, q0, hi, ln);

  // normalize + write fp32 output [B,S,D]
  #pragma unroll
  for (int sm = 0; sm < 2; ++sm)
    #pragma unroll
    for (int v = 0; v < 8; ++v) {
      const int   qr  = q0 + sm * 16 + v + hi * 8;
      const float inv = 1.0f / lrow[sm][v];
      #pragma unroll
      for (int t = 0; t < 4; ++t)
        out[((size_t)b * S_ + qr) * D_ + h * HD_ + t * 16 + ln] = acc[sm][t][v] * inv;
    }
}

// ---------------------------------------------------------------------------
extern "C" void kernel_launch(void* const* d_in, const int* in_sizes, int n_in,
                              void* d_out, int out_size, void* d_ws, size_t ws_size,
                              hipStream_t stream) {
  const float* x  = (const float*)d_in[0];
  const float* Wq = (const float*)d_in[1];
  const float* Wk = (const float*)d_in[2];
  const float* Wv = (const float*)d_in[3];
  float* out = (float*)d_out;

  const size_t xelems = (size_t)M_TOT * D_;        // 8Mi
  const size_t welems = (size_t)D_ * D_;           // 1Mi per projection
  const size_t qelems = (size_t)B_ * H_ * S_ * HD_;// 8Mi

  _Float16* xh = (_Float16*)d_ws;                  // [8192,1024]
  _Float16* wh = xh + xelems;                      // [3072,1024] (q,k,v rows)
  _Float16* Qh = wh + 3 * welems;                  // [B,H,S,HD]
  _Float16* Kh = Qh + qelems;                      // [B,H,S,HD]
  _Float16* Vt = Kh + qelems;                      // [B,H,HD,S]

  // 0) one-time fp32 -> f16 conversions
  cvt_f16_kernel<<<(int)(xelems / 8 / 256), 256, 0, stream>>>(x, xh, (int)(xelems / 8));
  cvt_f16_kernel<<<(int)(welems / 8 / 256), 256, 0, stream>>>(Wq, wh,              (int)(welems / 8));
  cvt_f16_kernel<<<(int)(welems / 8 / 256), 256, 0, stream>>>(Wk, wh + welems,     (int)(welems / 8));
  cvt_f16_kernel<<<(int)(welems / 8 / 256), 256, 0, stream>>>(Wv, wh + 2 * welems, (int)(welems / 8));

  // 1) QKV projection GEMM: (8192/128) x (3072/128) = 1536 blocks
  qkv_gemm_kernel<<<(M_TOT / BM) * (N_TOT / BN), 256, 0, stream>>>(xh, wh, Qh, Kh, Vt);

  // 2) flash attention: B*H*(S/32) = 4096 wave tiles, 4 waves/block
  flash_attn_kernel<<<4096 / 4, 128, 0, stream>>>(Qh, Kh, Vt, out);
}
